// loss2_35038343201101
// MI455X (gfx1250) — matrix operand order repaired and checked by reading, not compiled, not run on hardware
//
#include <hip/hip_runtime.h>
#include <hip/hip_bf16.h>

typedef __attribute__((ext_vector_type(16))) _Float16 v16h;
typedef __attribute__((ext_vector_type(8)))  _Float16 v8h;
typedef __attribute__((ext_vector_type(4)))  _Float16 v4h;
typedef __attribute__((ext_vector_type(8)))  float    v8f;
typedef __attribute__((ext_vector_type(4)))  float    v4f;

#define T_     4000
#define C_     128
#define BLK_   5
#define B_     (T_ / BLK_)      // 800 blocks
#define NNEG_  (2 * B_ - 2)     // 1598 negatives per pair
#define NPAIR_ (BLK_ * (BLK_ - 1)) // 20 ordered in-block pairs
#define NTILE_ (T_ / 16)        // 250 16-row tiles (4000 = 250*16 exactly)

// ---------------------------------------------------------------------------
// Stage 0: zero the scalar output (loss accumulated via atomics)
// ---------------------------------------------------------------------------
__global__ void k_zero(float* out) {
    if (blockIdx.x == 0 && threadIdx.x == 0) out[0] = 0.0f;
}

// ---------------------------------------------------------------------------
// Stage 1: L2-normalize each row, f32 -> f16.  One wave per row:
// 32 lanes x 4 floats = 128 columns; shfl_xor tree reduction for ||x||^2.
// ---------------------------------------------------------------------------
__global__ void k_normalize(const float* __restrict__ x,
                            _Float16* __restrict__ xn) {
    const int wave = (blockIdx.x * blockDim.x + threadIdx.x) >> 5;
    const int lane = threadIdx.x & 31;
    if (wave >= T_) return;

    const float* row = x + (size_t)wave * C_;
    v4f v = *(const v4f*)(row + lane * 4);
    float ss = v.x * v.x + v.y * v.y + v.z * v.z + v.w * v.w;
#pragma unroll
    for (int m = 16; m >= 1; m >>= 1) ss += __shfl_xor(ss, m, 32);

    const float inv = 1.0f / fmaxf(sqrtf(ss), 1e-8f);
    v4h o;
    o[0] = (_Float16)(v.x * inv);
    o[1] = (_Float16)(v.y * inv);
    o[2] = (_Float16)(v.z * inv);
    o[3] = (_Float16)(v.w * inv);
    *(v4h*)(xn + (size_t)wave * C_ + lane * 4) = o;
}

// ---------------------------------------------------------------------------
// Stage 2: fused  S = exp(Xn . Xn^T)  row-sum / block-sum.
// One wave per 16-row anchor tile.  For each of 250 column tiles:
//   4 x v_wmma_f32_16x16x32_f16 accumulate the full K=128 dot products,
//   epilogue applies exp and accumulates per-row sums (and the 5-wide
//   block-diagonal sum) without ever materializing the 64 MB sim matrix.
//
// Fragment layouts (wave32; A and B share this contiguous K order):
//   A (16x32 f16): lane M = lane&15; halves 0..15 -> K = 16*hi + t
//     => one contiguous 32B chunk of the anchor row.
//   B (32x16 f16): lane N = lane&15; halves 0..15 -> K = 16*hi + t
//     => one contiguous 32B chunk of row (ctBase + N) of Xn  (B == Xn^T).
//   C/D (16x16 f32): vgpr v, lane: M = v + 8*hi, N = lane&15.
// ---------------------------------------------------------------------------
__global__ void k_gram_rowsums(const _Float16* __restrict__ xn,
                               float* __restrict__ rowsum,
                               float* __restrict__ blocksum) {
    const int lane = threadIdx.x & 31;
    const int rt   = blockIdx.x * (blockDim.x >> 5) + (threadIdx.x >> 5);
    if (rt >= NTILE_) return;                 // uniform per wave: EXEC all-1s

    const int hi = lane >> 4;                 // half-wave select
    const int ln = lane & 15;

    // A fragments for all 4 K-steps (invariant over the column loop)
    const _Float16* rowA = xn + (size_t)(rt * 16 + ln) * C_;
    v16h afrag[4];
#pragma unroll
    for (int ks = 0; ks < 4; ++ks) {
        const int kb = ks * 32;
        afrag[ks] = *(const v16h*)(rowA + kb + 16 * hi);
    }

    float rowAcc[8], blkAcc[8];
    int   mBlk[8];
#pragma unroll
    for (int v = 0; v < 8; ++v) {
        rowAcc[v] = 0.0f;
        blkAcc[v] = 0.0f;
        mBlk[v]   = (rt * 16 + v + 8 * hi) / BLK_;   // block id of row M
    }

    for (int ct = 0; ct < NTILE_; ++ct) {
        const _Float16* rowB = xn + (size_t)(ct * 16 + ln) * C_;
        if (ct + 1 < NTILE_)
            __builtin_prefetch(xn + (size_t)((ct + 1) * 16 + ln) * C_, 0, 0);

        v8f c = {};
#pragma unroll
        for (int ks = 0; ks < 4; ++ks) {
            v16h b = *(const v16h*)(rowB + ks * 32 + 16 * hi);
            c = __builtin_amdgcn_wmma_f32_16x16x32_f16(
                    false, afrag[ks], false, b, (short)0, c, false, false);
        }

        const int nBlk = (ct * 16 + ln) / BLK_;      // block id of column N
#pragma unroll
        for (int v = 0; v < 8; ++v) {
            const float e = __expf(c[v]);
            rowAcc[v] += e;
            if (nBlk == mBlk[v]) blkAcc[v] += e;     // in-block (incl. self)
        }
    }

    // Reduce over the 16 lanes of each half-wave (N dimension)
#pragma unroll
    for (int m = 8; m >= 1; m >>= 1) {
#pragma unroll
        for (int v = 0; v < 8; ++v) {
            rowAcc[v] += __shfl_xor(rowAcc[v], m, 32);
            blkAcc[v] += __shfl_xor(blkAcc[v], m, 32);
        }
    }
    if (ln == 0) {                                   // lanes 0 and 16 write
#pragma unroll
        for (int v = 0; v < 8; ++v) {
            const int r = rt * 16 + v + 8 * hi;
            rowsum[r]   = rowAcc[v];
            blocksum[r] = blkAcc[v];
        }
    }
}

// ---------------------------------------------------------------------------
// Stage 3: per-pair loss.  pos = exp(xn_i . xn_j); negatives are a uniform
// random 1598-of-3995 subset -> expected sum = (rowsum - blocksum) * 1598/3995.
// Wave-level reduction, then one atomicAdd per wave into the scalar loss.
// ---------------------------------------------------------------------------
__global__ void k_loss(const _Float16* __restrict__ xn,
                       const float* __restrict__ rowsum,
                       const float* __restrict__ blocksum,
                       const int* __restrict__ kuai2p,
                       float* __restrict__ out) {
    const int tid = blockIdx.x * blockDim.x + threadIdx.x;
    float term = 0.0f;
    if (tid < B_ * NPAIR_) {
        const int b = tid / NPAIR_;
        const int p = tid % NPAIR_;
        const int pi = p / (BLK_ - 1);
        int       pj = p % (BLK_ - 1);
        if (pj >= pi) ++pj;                          // skip i == j
        const int i = b * BLK_ + pi;
        const int j = b * BLK_ + pj;

        const _Float16* ri = xn + (size_t)i * C_;
        const _Float16* rj = xn + (size_t)j * C_;
        float dot = 0.0f;
#pragma unroll 8
        for (int k = 0; k < C_; ++k) dot += (float)ri[k] * (float)rj[k];

        const float pos    = __expf(dot);
        const float cand   = rowsum[i] - blocksum[i];          // excl. block
        const float negsum = cand * ((float)NNEG_ / (float)(T_ - BLK_));
        const int   kuai2  = kuai2p[0];
        const float denom  = (float)B_ * (float)kuai2 * (float)(kuai2 - 1);
        term = -__logf(pos / (negsum + pos)) / denom;
    }
#pragma unroll
    for (int m = 16; m >= 1; m >>= 1) term += __shfl_xor(term, m, 32);
    if ((threadIdx.x & 31) == 0) atomicAdd(out, term);
}

// ---------------------------------------------------------------------------
extern "C" void kernel_launch(void* const* d_in, const int* in_sizes, int n_in,
                              void* d_out, int out_size, void* d_ws, size_t ws_size,
                              hipStream_t stream) {
    const float* x      = (const float*)d_in[0];
    const int*   kuai2  = (const int*)d_in[1];
    float*       out    = (float*)d_out;

    char* ws = (char*)d_ws;
    _Float16* xn       = (_Float16*)ws;                       // 4000*128*2 = 1,024,000 B
    float*    rowsum   = (float*)(ws + (1u << 20));           // 16,000 B
    float*    blocksum = rowsum + T_;                         // 16,000 B

    k_zero<<<1, 32, 0, stream>>>(out);

    // 4000 rows, 8 waves (256 threads) per block -> 500 blocks
    k_normalize<<<(T_ + 7) / 8, 256, 0, stream>>>(x, xn);

    // 250 row tiles, 4 waves (128 threads) per block -> 63 blocks
    k_gram_rowsums<<<(NTILE_ + 3) / 4, 128, 0, stream>>>(xn, rowsum, blocksum);

    // 800 * 20 = 16000 pair terms
    const int npair = B_ * NPAIR_;
    k_loss<<<(npair + 255) / 256, 256, 0, stream>>>(xn, rowsum, blocksum, kuai2, out);
}
